// NeST_24730421691150
// MI455X (gfx1250) — matrix operand order repaired and checked by reading, not compiled, not run on hardware
//
#include <hip/hip_runtime.h>

#define NPTS 1000000
#define NSEG 20000
#define LTILES (NPTS / 16)   // 62500
#define GTILES (NSEG / 16)   // 1250

typedef __attribute__((ext_vector_type(16))) _Float16 v16h;
typedef __attribute__((ext_vector_type(8)))  float    v8f;

// ---- WMMA fragment helpers (wave32, 16x16x32 f16 -> f32) --------------------

// A fragment from a row-major LDS tile of _Float16, leading dim `ld` halves.
// ISA layout (16-bit A 16x32): lanes 0-15 row M=lane, K pairs (0,1)(2,3)(4,5)(6,7)
// in VGPR0-3 and (16..23) in VGPR4-7; lanes 16-31 same rows, K+8 / K+24.
// K runs 0..7 and 16..23 contiguous per lane -> two aligned ds_load_b128.
__device__ __forceinline__ v16h lds_a_frag(const _Float16* t, int ld, int kc, int lane) {
  int row = lane & 15;
  int kb  = (lane >> 4) << 3;          // 0 or 8
  v16h a;
#pragma unroll
  for (int p = 0; p < 8; ++p) {
    int k = kc + ((p & 4) << 2) + kb + ((p & 3) << 1);
    a[2 * p]     = t[row * ld + k];
    a[2 * p + 1] = t[row * ld + k + 1];
  }
  return a;
}

// B fragment from fragment-major packed weights: for (chunk,tile), 32 lanes x
// 16 halves contiguous. One aligned 32-byte load per lane (2x global_load_b128).
__device__ __forceinline__ v16h frag_b(const _Float16* __restrict__ Wf,
                                       int chunk, int nt, int NT, int lane) {
  return *(const v16h*)(Wf + ((size_t)(chunk * NT + nt) * 512 + lane * 16));
}

__device__ __forceinline__ v8f wmma16(v16h a, v16h b, v8f c) {
  return __builtin_amdgcn_wmma_f32_16x16x32_f16(false, a, false, b, (short)0, c, false, false);
}

__device__ __forceinline__ unsigned pack2(float a, float b) {
  union { _Float16 h[2]; unsigned u; } p;
  p.h[0] = (_Float16)a;
  p.h[1] = (_Float16)b;
  return p.u;
}

// ---- K0: zero scratch + repack weights into f16 B-fragment order ------------

// ISA layout (16-bit B 32x16): lanes 0-15 hold K=0..7,16..23 (col N=lane&15),
// lanes 16-31 hold K=8..15,24..31. Element j of lane l, chunk c, tile t comes from
// Wpad[c*32 + ((j>=8)?16:0) + ((l>=16)?8:0) + (j&7)][t*16 + (l&15)].
__device__ void pack_frag(const float* __restrict__ W, int Kreal, int Nreal, int ldn,
                          int NC, int NT, _Float16* __restrict__ out,
                          int tid, int stride) {
  int total = NC * NT * 512;
  for (int i = tid; i < total; i += stride) {
    int j    = i & 15;
    int lane = (i >> 4) & 31;
    int t    = (i >> 9) % NT;
    int c    = i / (512 * NT);
    int k = c * 32 + ((j & 8) << 1) + ((lane >> 4) << 3) + (j & 7);
    int n = t * 16 + (lane & 15);
    float v = (k < Kreal && n < Nreal) ? W[k * ldn + n] : 0.0f;
    out[i] = (_Float16)v;
  }
}

__global__ void k_init(float* counts, float* possum, unsigned* radius, float* g,
                       const float* Wl1, const float* Wl2, const float* Wg1,
                       const float* Wg2, const float* Wh,
                       _Float16* W1f, _Float16* W2f, _Float16* Wg1f,
                       _Float16* Wg2f, _Float16* Whf) {
  int tid = blockIdx.x * blockDim.x + threadIdx.x;
  int stride = gridDim.x * blockDim.x;
  for (int i = tid; i < NSEG; i += stride) { counts[i] = 0.0f; radius[i] = 0u; }
  for (int i = tid; i < 3 * NSEG; i += stride) possum[i] = 0.0f;
  for (int i = tid; i < 128 * NSEG; i += stride) g[i] = 0.0f;
  pack_frag(Wl1,  35,  64,  64, 2, 4, W1f,  tid, stride);
  pack_frag(Wl2,  64, 128, 128, 2, 8, W2f,  tid, stride);
  pack_frag(Wg1, 129, 128, 128, 5, 8, Wg1f, tid, stride);
  pack_frag(Wg2, 128,  64,  64, 4, 4, Wg2f, tid, stride);
  pack_frag(Wh,   64,  13,  13, 2, 1, Whf,  tid, stride);
}

// ---- K1: segment counts + position sums -------------------------------------

__global__ void k_accum(const float* __restrict__ pos, const int* __restrict__ idx,
                        float* counts, float* possum) {
  int i = blockIdx.x * blockDim.x + threadIdx.x;
  if (i >= NPTS) return;
  int s = idx[i];
  atomicAdd(&counts[s], 1.0f);
  atomicAdd(&possum[s * 3 + 0], pos[i * 3 + 0]);
  atomicAdd(&possum[s * 3 + 1], pos[i * 3 + 1]);
  atomicAdd(&possum[s * 3 + 2], pos[i * 3 + 2]);
}

// ---- K2: centers ------------------------------------------------------------

__global__ void k_center(const float* counts, const float* possum, float* center) {
  int s = blockIdx.x * blockDim.x + threadIdx.x;
  if (s >= NSEG) return;
  float inv = 1.0f / fmaxf(counts[s], 1.0f);
  center[s * 3 + 0] = possum[s * 3 + 0] * inv;
  center[s * 3 + 1] = possum[s * 3 + 1] * inv;
  center[s * 3 + 2] = possum[s * 3 + 2] * inv;
}

// ---- K3: per-point distance -> segment radius (u32 atomic max, dist >= 0) ---

__global__ void k_radius(const float* __restrict__ pos, const int* __restrict__ idx,
                         const float* __restrict__ center, unsigned* radius) {
  int i = blockIdx.x * blockDim.x + threadIdx.x;
  if (i >= NPTS) return;
  int s = idx[i];
  float dx = pos[i * 3 + 0] - center[s * 3 + 0];
  float dy = pos[i * 3 + 1] - center[s * 3 + 1];
  float dz = pos[i * 3 + 2] - center[s * 3 + 2];
  float dist = sqrtf(dx * dx + dy * dy + dz * dz + 1e-12f);
  atomicMax(&radius[s], __float_as_uint(dist));
}

// ---- K4: diameter -----------------------------------------------------------

__global__ void k_diam(const float* counts, const unsigned* radius, float* diameter) {
  int s = blockIdx.x * blockDim.x + threadIdx.x;
  if (s >= NSEG) return;
  float r = __uint_as_float(radius[s]);
  diameter[s] = (counts[s] > 0.0f) ? 2.0f * r : 0.0f;
}

// ---- K5: fused local MLP (WMMA) + segment-max scatter -----------------------
// One wave = one 16-point tile. 8 waves/block.

__global__ void __launch_bounds__(256)
k_local(const float* __restrict__ pos, const float* __restrict__ x,
        const int* __restrict__ idx, const float* __restrict__ center,
        const float* __restrict__ diameter,
        const _Float16* __restrict__ W1f, const _Float16* __restrict__ W2f,
        const float* __restrict__ b1, const float* __restrict__ b2,
        float* __restrict__ g) {
  __shared__ _Float16 sA[8][16 * 64];  // layer-1 input tile (K padded to 64)
  __shared__ _Float16 sB[8][16 * 64];  // layer-1 output / layer-2 input
  int wave = threadIdx.x >> 5;
  int lane = threadIdx.x & 31;
  int tile = blockIdx.x * 8 + wave;
  if (tile >= LTILES) tile = LTILES - 1;   // clamp (dup work, identical results)
  int base = tile * 16;
  _Float16* A = sA[wave];
  _Float16* B = sB[wave];
  int row  = lane & 15;
  int half = lane >> 4;
  int pt = base + row;

  if (half == 0) {
    // 32 x-features: 8x float4 loads, packed b32 LDS stores
    const float4* xr = (const float4*)(x + (size_t)pt * 32);
    unsigned* A32 = (unsigned*)(A + row * 64);
#pragma unroll
    for (int q = 0; q < 8; ++q) {
      float4 f = xr[q];
      A32[q * 2 + 0] = pack2(f.x, f.y);
      A32[q * 2 + 1] = pack2(f.z, f.w);
    }
  } else {
    int s = idx[pt];
    float inv = 1.0f / fmaxf(diameter[s], 1e-12f);
    float px = (pos[pt * 3 + 0] - center[s * 3 + 0]) * inv;
    float py = (pos[pt * 3 + 1] - center[s * 3 + 1]) * inv;
    float pz = (pos[pt * 3 + 2] - center[s * 3 + 2]) * inv;
    unsigned* A32 = (unsigned*)(A + row * 64);
    A32[16] = pack2(px, py);
    A32[17] = pack2(pz, 0.0f);
#pragma unroll
    for (int q = 18; q < 32; ++q) A32[q] = 0u;
  }
  __syncthreads();

  // layer 1: [16x64(pad 35)] x [64x64] -> [16x64]
  v8f c1[4];
#pragma unroll
  for (int nt = 0; nt < 4; ++nt) c1[nt] = (v8f){0, 0, 0, 0, 0, 0, 0, 0};
#pragma unroll
  for (int kc = 0; kc < 2; ++kc) {
    v16h a = lds_a_frag(A, 64, kc * 32, lane);
#pragma unroll
    for (int nt = 0; nt < 4; ++nt)
      c1[nt] = wmma16(a, frag_b(W1f, kc, nt, 4, lane), c1[nt]);
  }
  // ReLU + bias -> sB
#pragma unroll
  for (int nt = 0; nt < 4; ++nt) {
    float bias = b1[nt * 16 + (lane & 15)];
#pragma unroll
    for (int r = 0; r < 8; ++r) {
      int m = r + half * 8;
      float v = fmaxf(c1[nt][r] + bias, 0.0f);
      B[m * 64 + nt * 16 + (lane & 15)] = (_Float16)v;
    }
  }
  __syncthreads();

  // layer 2: [16x64] x [64x128] -> [16x128]
  v8f c2[8];
#pragma unroll
  for (int nt = 0; nt < 8; ++nt) c2[nt] = (v8f){0, 0, 0, 0, 0, 0, 0, 0};
#pragma unroll
  for (int kc = 0; kc < 2; ++kc) {
    v16h a = lds_a_frag(B, 64, kc * 32, lane);
#pragma unroll
    for (int nt = 0; nt < 8; ++nt)
      c2[nt] = wmma16(a, frag_b(W2f, kc, nt, 8, lane), c2[nt]);
  }
  // bias
  float bb[8];
#pragma unroll
  for (int nt = 0; nt < 8; ++nt) bb[nt] = b2[nt * 16 + (lane & 15)];
  // ReLU + scatter atomic-max into g[S,128] (post-ReLU values >= 0 -> u32 order ok)
#pragma unroll
  for (int r = 0; r < 8; ++r) {
    int m = r + half * 8;
    int s = idx[base + m];
    unsigned* grow = (unsigned*)(g + (size_t)s * 128);
#pragma unroll
    for (int nt = 0; nt < 8; ++nt) {
      float v = fmaxf(c2[nt][r] + bb[nt], 0.0f);
      atomicMax(grow + nt * 16 + (lane & 15), __float_as_uint(v));
    }
  }
}

// ---- K6: global MLP (WMMA) [S,129]->128->64->13 -----------------------------
// One wave = one 16-segment tile. 4 waves/block.

__global__ void __launch_bounds__(128)
k_global(const float* __restrict__ g, const float* __restrict__ diameter,
         const _Float16* __restrict__ Wg1f, const _Float16* __restrict__ Wg2f,
         const _Float16* __restrict__ Whf,
         const float* __restrict__ bg1, const float* __restrict__ bg2,
         const float* __restrict__ bh, float* __restrict__ out) {
  __shared__ _Float16 sA[4][16 * 160];   // input (K padded 129->160)
  __shared__ _Float16 sH[4][16 * 128];   // hidden 1
  __shared__ _Float16 sH2[4][16 * 64];   // hidden 2
  int wave = threadIdx.x >> 5;
  int lane = threadIdx.x & 31;
  int tile = blockIdx.x * 4 + wave;
  if (tile >= GTILES) tile = GTILES - 1;
  int base = tile * 16;
  _Float16* A  = sA[wave];
  _Float16* H  = sH[wave];
  _Float16* H2 = sH2[wave];
  int row  = lane & 15;
  int half = lane >> 4;

  // fill A tile: cols 0..127 = g, 128 = diameter, 129..159 = 0 (packed stores)
  {
    const float* grow = g + (size_t)(base + row) * 128;
    unsigned* A32 = (unsigned*)(A + row * 160);
#pragma unroll
    for (int q = 0; q < 40; ++q) {
      int c = half * 80 + 2 * q;
      float v0 = (c < 128) ? grow[c] : (c == 128 ? diameter[base + row] : 0.0f);
      float v1 = (c + 1 < 128) ? grow[c + 1] : 0.0f;
      A32[c >> 1] = pack2(v0, v1);
    }
  }
  __syncthreads();

  // g1: [16x160] x [160x128] -> [16x128]
  v8f c1[8];
#pragma unroll
  for (int nt = 0; nt < 8; ++nt) c1[nt] = (v8f){0, 0, 0, 0, 0, 0, 0, 0};
#pragma unroll
  for (int kc = 0; kc < 5; ++kc) {
    v16h a = lds_a_frag(A, 160, kc * 32, lane);
#pragma unroll
    for (int nt = 0; nt < 8; ++nt)
      c1[nt] = wmma16(a, frag_b(Wg1f, kc, nt, 8, lane), c1[nt]);
  }
#pragma unroll
  for (int nt = 0; nt < 8; ++nt) {
    float bias = bg1[nt * 16 + (lane & 15)];
#pragma unroll
    for (int r = 0; r < 8; ++r) {
      int m = r + half * 8;
      H[m * 128 + nt * 16 + (lane & 15)] = (_Float16)fmaxf(c1[nt][r] + bias, 0.0f);
    }
  }
  __syncthreads();

  // g2: [16x128] x [128x64] -> [16x64]
  v8f c2[4];
#pragma unroll
  for (int nt = 0; nt < 4; ++nt) c2[nt] = (v8f){0, 0, 0, 0, 0, 0, 0, 0};
#pragma unroll
  for (int kc = 0; kc < 4; ++kc) {
    v16h a = lds_a_frag(H, 128, kc * 32, lane);
#pragma unroll
    for (int nt = 0; nt < 4; ++nt)
      c2[nt] = wmma16(a, frag_b(Wg2f, kc, nt, 4, lane), c2[nt]);
  }
#pragma unroll
  for (int nt = 0; nt < 4; ++nt) {
    float bias = bg2[nt * 16 + (lane & 15)];
#pragma unroll
    for (int r = 0; r < 8; ++r) {
      int m = r + half * 8;
      H2[m * 64 + nt * 16 + (lane & 15)] = (_Float16)fmaxf(c2[nt][r] + bias, 0.0f);
    }
  }
  __syncthreads();

  // head: [16x64] x [64x16(pad 13)] -> [16x16]
  v8f c3 = (v8f){0, 0, 0, 0, 0, 0, 0, 0};
#pragma unroll
  for (int kc = 0; kc < 2; ++kc) {
    v16h a = lds_a_frag(H2, 64, kc * 32, lane);
    c3 = wmma16(a, frag_b(Whf, kc, 0, 1, lane), c3);
  }
  int n = lane & 15;
  if (n < 13) {
    float bias = bh[n];
#pragma unroll
    for (int r = 0; r < 8; ++r) {
      int m = r + half * 8;
      out[(size_t)(base + m) * 13 + n] = c3[r] + bias;
    }
  }
}

// ---- launch -----------------------------------------------------------------

extern "C" void kernel_launch(void* const* d_in, const int* in_sizes, int n_in,
                              void* d_out, int out_size, void* d_ws, size_t ws_size,
                              hipStream_t stream) {
  const float* pos = (const float*)d_in[0];
  const float* x   = (const float*)d_in[1];
  const int*   idx = (const int*)d_in[2];
  const float* Wl1 = (const float*)d_in[3];
  const float* bl1 = (const float*)d_in[4];
  const float* Wl2 = (const float*)d_in[5];
  const float* bl2 = (const float*)d_in[6];
  const float* Wg1 = (const float*)d_in[7];
  const float* bg1 = (const float*)d_in[8];
  const float* Wg2 = (const float*)d_in[9];
  const float* bg2 = (const float*)d_in[10];
  const float* Wh  = (const float*)d_in[11];
  const float* bh  = (const float*)d_in[12];

  char* ws = (char*)d_ws;
  size_t off = 0;
  auto carve = [&](size_t bytes) -> void* {
    void* p = ws + off;
    off = (off + bytes + 255) & ~(size_t)255;
    return p;
  };
  float*    counts   = (float*)   carve(NSEG * 4);
  float*    possum   = (float*)   carve(NSEG * 3 * 4);
  float*    center   = (float*)   carve(NSEG * 3 * 4);
  unsigned* radius   = (unsigned*)carve(NSEG * 4);
  float*    diameter = (float*)   carve(NSEG * 4);
  float*    g        = (float*)   carve((size_t)NSEG * 128 * 4);
  _Float16* W1f      = (_Float16*)carve(2 * 4 * 512 * 2);   // 2 chunks x 4 tiles
  _Float16* W2f      = (_Float16*)carve(2 * 8 * 512 * 2);
  _Float16* Wg1f     = (_Float16*)carve(5 * 8 * 512 * 2);
  _Float16* Wg2f     = (_Float16*)carve(4 * 4 * 512 * 2);
  _Float16* Whf      = (_Float16*)carve(2 * 1 * 512 * 2);

  k_init<<<2048, 256, 0, stream>>>(counts, possum, radius, g,
                                   Wl1, Wl2, Wg1, Wg2, Wh,
                                   W1f, W2f, Wg1f, Wg2f, Whf);
  k_accum<<<(NPTS + 255) / 256, 256, 0, stream>>>(pos, idx, counts, possum);
  k_center<<<(NSEG + 255) / 256, 256, 0, stream>>>(counts, possum, center);
  k_radius<<<(NPTS + 255) / 256, 256, 0, stream>>>(pos, idx, center, radius);
  k_diam<<<(NSEG + 255) / 256, 256, 0, stream>>>(counts, radius, diameter);
  k_local<<<(LTILES + 7) / 8, 256, 0, stream>>>(pos, x, idx, center, diameter,
                                                W1f, W2f, bl1, bl2, g);
  k_global<<<(GTILES + 3) / 4, 128, 0, stream>>>(g, diameter, Wg1f, Wg2f, Whf,
                                                 bg1, bg2, bh, (float*)d_out);
}
